// SemGCN_FC_5368709120324
// MI455X (gfx1250) — matrix operand order repaired and checked by reading, not compiled, not run on hardware
//
#include <hip/hip_runtime.h>
#include <hip/hip_bf16.h>
#include <math.h>

typedef __attribute__((ext_vector_type(16))) __bf16        v16bf;
typedef __attribute__((ext_vector_type(8)))  float         v8f;
typedef __attribute__((ext_vector_type(8)))  unsigned int  v8u;
typedef __attribute__((ext_vector_type(4)))  unsigned int  u32x4;

#define NJ     21
#define NBATCH 4096
#define NROWS  (NBATCH * NJ)   /* 86016, divisible by 128 and by 256 */

/* neighbor lists (excluding self), ascending, from the fixed EDGES graph */
__device__ __constant__ int c_nnb[NJ] = {1,3,2,2,1, 3,2,2,1, 3,2,2,1, 3,2,2,1, 2,2,2,1};
__device__ __constant__ int c_nbr[NJ][3] = {
    {1,-1,-1},{0,2,5},{1,3,-1},{2,4,-1},{3,-1,-1},
    {1,6,9},{5,7,-1},{6,8,-1},{7,-1,-1},
    {5,10,13},{9,11,-1},{10,12,-1},{11,-1,-1},
    {9,14,17},{13,15,-1},{14,16,-1},{15,-1,-1},
    {13,18,-1},{17,19,-1},{18,20,-1},{19,-1,-1}};

/* ---- adjacency: scatter e (np.nonzero row-major order) + row softmax ---- */
__global__ void adj_softmax_kernel(const float* __restrict__ e, float* __restrict__ adj)
{
    if (threadIdx.x != 0) return;
    for (int i = 0; i < NJ * NJ; ++i) adj[i] = 0.0f;
    int k = 0;
    for (int i = 0; i < NJ; ++i) {
        float ev[4]; int cols[4]; int cc = 0; bool placed = false;
        int nb = c_nnb[i];
        for (int t = 0; t < nb; ++t) {
            int j = c_nbr[i][t];
            if (!placed && i < j) { cols[cc] = i; ev[cc] = e[k++]; ++cc; placed = true; }
            cols[cc] = j; ev[cc] = e[k++]; ++cc;
        }
        if (!placed) { cols[cc] = i; ev[cc] = e[k++]; ++cc; }
        float m = -1e30f;
        for (int t = 0; t < cc; ++t) m = fmaxf(m, ev[t]);
        float s = 0.0f; float ex[4];
        for (int t = 0; t < cc; ++t) { ex[t] = expf(ev[t] - m); s += ex[t]; }
        float inv = 1.0f / s;
        for (int t = 0; t < cc; ++t) adj[i * NJ + cols[t]] = ex[t] * inv;
    }
}

/* pack two fp32 into one dword of two bf16 */
__device__ __forceinline__ unsigned pack_bf16(float f0, float f1)
{
    __bf16 b0 = (__bf16)f0;
    __bf16 b1 = (__bf16)f1;
    return ((unsigned)__builtin_bit_cast(unsigned short, b1) << 16)
         |  (unsigned)__builtin_bit_cast(unsigned short, b0);
}

struct u32x8s { u32x4 lo, hi; };

/* ---- dual GEMM: H0 = A*W0, H1 = A*W1 via v_wmma_f32_16x16x32_bf16 ----
 * block = 256 thr (8 wave32), tile = 128(M) x 64(N), K in steps of 32.
 * LDS is laid out fragment-contiguously (row stride 20 dwords = 80B:
 * 16B-aligned halves for ds_load_b128, and 20n mod 64 distinct for
 * n=0..15 so half-wave lanes hit distinct banks).  Double buffered:
 * one barrier per K step; global loads of tile k+1 overlap WMMAs of k. */
__global__ __launch_bounds__(256)
void wmma_dual_gemm(const float* __restrict__ A,
                    const float* __restrict__ W0,
                    const float* __restrict__ W1,
                    float* __restrict__ H0,
                    float* __restrict__ H1,
                    int K, int N)
{
    __shared__ __align__(16) unsigned ldsA[2][128][20];    /* [buf][row][kk]      */
    __shared__ __align__(16) unsigned ldsB[2][2][64][20];  /* [buf][mat][col][kk] */

    const int tid  = threadIdx.x;
    const int wave = tid >> 5;
    const int lane = tid & 31;
    const int hb   = lane >> 4;             /* half-wave */
    const int l16  = lane & 15;
    const int m0   = blockIdx.x * 128;
    const int n0   = blockIdx.y * 64;
    const float* Wm[2] = { W0, W1 };

    v8f acc[2][4];
    v8f zero = {};
    for (int s = 0; s < 2; ++s)
        for (int g = 0; g < 4; ++g) acc[s][g] = zero;

    const int nk = K >> 5;

    auto load_tile = [&](int kt, int bsel) {
        /* A tile 128x32 fp32 -> 2048 bf16 pairs (coalesced float2 loads) */
        #pragma unroll
        for (int t = 0; t < 8; ++t) {
            int p  = t * 256 + tid;
            int m  = p >> 4, kk = p & 15;
            const float* src = A + (size_t)(m0 + m) * K + kt + kk * 2;
            float2 fv = *(const float2*)src;
            ldsA[bsel][m][kk] = pack_bf16(fv.x, fv.y);
            if (kt + 64 < K) __builtin_prefetch(src + 64, 0, 1);  /* global_prefetch_b8 */
        }
        /* W0/W1 tiles 32x64 each -> 2048 pairs, stored col-major per lane */
        #pragma unroll
        for (int t = 0; t < 8; ++t) {
            int q = t * 256 + tid;
            int s = q >> 10, rem = q & 1023;
            int kk = rem >> 6, n = rem & 63;
            const float* w = Wm[s] + (size_t)(kt + kk * 2) * N + n0 + n;
            ldsB[bsel][s][n][kk] = pack_bf16(w[0], w[N]);
        }
    };

    load_tile(0, 0);
    __syncthreads();

    int buf = 0;
    for (int ki = 0; ki < nk; ++ki) {
        if (ki + 1 < nk) load_tile((ki + 1) << 5, buf ^ 1);

        /* A fragment: two contiguous b128 loads per lane (ISA 7.12.2 layout) */
        const unsigned* ap = &ldsA[buf][wave * 16 + l16][0];
        u32x8s ua;
        ua.lo = *(const u32x4*)(ap + hb * 4);
        ua.hi = *(const u32x4*)(ap + 8 + hb * 4);
        v16bf afrag = __builtin_bit_cast(v16bf, ua);

        #pragma unroll
        for (int s = 0; s < 2; ++s) {
            #pragma unroll
            for (int g = 0; g < 4; ++g) {
                const unsigned* bp = &ldsB[buf][s][g * 16 + l16][0];
                u32x8s ub;
                ub.lo = *(const u32x4*)(bp + hb * 4);
                ub.hi = *(const u32x4*)(bp + 8 + hb * 4);
                v16bf bfrag = __builtin_bit_cast(v16bf, ub);
                acc[s][g] = __builtin_amdgcn_wmma_f32_16x16x32_bf16(
                    false, afrag, false, bfrag, (short)0, acc[s][g], false, false);
            }
        }
        __syncthreads();
        buf ^= 1;
    }

    /* C/D layout: VGPR r -> row hb*8+r, lane&15 -> col */
    for (int s = 0; s < 2; ++s) {
        float* H = s ? H1 : H0;
        for (int g = 0; g < 4; ++g) {
            #pragma unroll
            for (int r = 0; r < 8; ++r) {
                int row = m0 + wave * 16 + hb * 8 + r;
                int col = n0 + g * 16 + l16;
                H[(size_t)row * N + col] = acc[s][g][r];
            }
        }
    }
}

/* ---- input layer: K=2, compute h0/h1 directly ---- */
__global__ void input_gemv_kernel(const float* __restrict__ X, const float* __restrict__ W,
                                  float* __restrict__ H0, float* __restrict__ H1)
{
    long long idx = (long long)blockIdx.x * blockDim.x + threadIdx.x;
    if (idx >= (long long)NROWS * 128) return;
    int d = (int)(idx & 127);
    long long row = idx >> 7;
    float x0 = X[row * 2], x1 = X[row * 2 + 1];
    const float* W0 = W;            /* (2,128) */
    const float* W1 = W + 256;
    H0[idx] = x0 * W0[d] + x1 * W0[128 + d];
    H1[idx] = x0 * W1[d] + x1 * W1[128 + d];
}

/* ---- graph combine: y = adj_ii*h0(i) + sum_nbr adj_ij*h1(j) + bias ---- */
__global__ void combine_kernel(const float* __restrict__ h0, const float* __restrict__ h1,
                               const float* __restrict__ adj, const float* __restrict__ bias,
                               float* __restrict__ y, int C, long long total)
{
    long long idx = (long long)blockIdx.x * blockDim.x + threadIdx.x;
    if (idx >= total) return;
    int d = (int)(idx % C);
    long long row = idx / C;
    int i = (int)(row % NJ);
    long long rb = row - i;                        /* b*21 */
    float v = adj[i * NJ + i] * h0[idx];
    int nb = c_nnb[i];
    for (int t = 0; t < nb; ++t) {
        int j = c_nbr[i][t];
        v += adj[i * NJ + j] * h1[(rb + j) * C + d];
    }
    y[idx] = v + bias[d];
}

/* ---- BN stats, deterministic two-stage ---- */
__global__ void bn_stats1(const float* __restrict__ y, float* __restrict__ psum,
                          float* __restrict__ psq, int C)
{
    int blk = blockIdx.x;                    /* 336 blocks x 256 rows */
    long long rowBase = (long long)blk * 256;
    for (int c = threadIdx.x; c < C; c += 256) {
        float s = 0.0f, q = 0.0f;
        for (int r = 0; r < 256; ++r) {
            float v = y[(rowBase + r) * C + c];
            s += v; q += v * v;
        }
        psum[(long long)blk * C + c] = s;
        psq [(long long)blk * C + c] = q;
    }
}

__global__ void bn_stats2(const float* __restrict__ psum, const float* __restrict__ psq,
                          const float* __restrict__ gamma, const float* __restrict__ beta,
                          float* __restrict__ scale, float* __restrict__ shift, int C)
{
    int c = blockIdx.x * blockDim.x + threadIdx.x;
    if (c >= C) return;
    float s = 0.0f, q = 0.0f;
    for (int b = 0; b < 336; ++b) { s += psum[(long long)b * C + c]; q += psq[(long long)b * C + c]; }
    const float inv_n = 1.0f / (float)NROWS;
    float mean = s * inv_n;
    float var  = q * inv_n - mean * mean;       /* biased, as reference */
    float sc = gamma[c] * rsqrtf(var + 1e-5f);
    scale[c] = sc;
    shift[c] = beta[c] - mean * sc;
}

__global__ void bn_apply(float* __restrict__ y, const float* __restrict__ scale,
                         const float* __restrict__ shift, int C, long long total)
{
    long long idx = (long long)blockIdx.x * blockDim.x + threadIdx.x;
    if (idx >= total) return;
    int c = (int)(idx % C);
    y[idx] = fmaxf(y[idx] * scale[c] + shift[c], 0.0f);
}

/* ---- residual add ---- */
__global__ void add_kernel(const float* __restrict__ a, const float* __restrict__ b,
                           float* __restrict__ z, long long total)
{
    long long idx = (long long)blockIdx.x * blockDim.x + threadIdx.x;
    if (idx >= total) return;
    z[idx] = a[idx] + b[idx];
}

/* ---- SE node attention ---- */
__global__ void smean_kernel(const float* __restrict__ z, float* __restrict__ s, int C)
{
    long long idx = (long long)blockIdx.x * blockDim.x + threadIdx.x;
    if (idx >= (long long)NBATCH * C) return;
    int c = (int)(idx % C);
    long long b = idx / C;
    long long base = b * NJ * C + c;
    float acc = 0.0f;
    for (int i = 0; i < NJ; ++i) acc += z[base + (long long)i * C];
    s[idx] = acc * (1.0f / (float)NJ);
}

__global__ void fc1_kernel(const float* __restrict__ s, const float* __restrict__ w,
                           const float* __restrict__ bias, float* __restrict__ o,
                           int C, int C4)
{
    long long idx = (long long)blockIdx.x * blockDim.x + threadIdx.x;
    if (idx >= (long long)NBATCH * C4) return;
    int oc = (int)(idx % C4);
    long long b = idx / C4;
    float acc = bias[oc];
    const float* sp = s + b * C;
    for (int k = 0; k < C; ++k) acc += sp[k] * w[(long long)k * C4 + oc];
    o[idx] = fmaxf(acc, 0.0f);
}

__global__ void fc2_kernel(const float* __restrict__ h, const float* __restrict__ w,
                           const float* __restrict__ bias, float* __restrict__ o, int C4)
{
    long long idx = (long long)blockIdx.x * blockDim.x + threadIdx.x;
    if (idx >= (long long)NBATCH * NJ) return;
    int i = (int)(idx % NJ);
    long long b = idx / NJ;
    float acc = bias[i];
    const float* hp = h + b * C4;
    for (int k = 0; k < C4; ++k) acc += hp[k] * w[k * NJ + i];
    o[idx] = 1.0f / (1.0f + expf(-acc));
}

__global__ void att_apply(const float* __restrict__ z, const float* __restrict__ w,
                          float* __restrict__ x, int C, long long total)
{
    long long idx = (long long)blockIdx.x * blockDim.x + threadIdx.x;
    if (idx >= total) return;
    long long row = idx / C;
    x[idx] = z[idx] * (1.0f + w[row]);
}

/* ---- concat: xout[b,i,:] = [ jf[b,:,i] (256) , xin[b,i,:] (Cin) ] ---- */
__global__ void concat_kernel(const float* __restrict__ jf, const float* __restrict__ xin,
                              float* __restrict__ xout, int Cin, int Cout)
{
    int b = blockIdx.x;
    const float* jb = jf + (long long)b * 256 * NJ;
    for (int t = threadIdx.x; t < 256 * NJ; t += blockDim.x) {
        int c = t / NJ, i = t % NJ;
        xout[((long long)b * NJ + i) * Cout + c] = jb[t];
    }
    for (int t = threadIdx.x; t < NJ * Cin; t += blockDim.x) {
        int i = t / Cin, c = t % Cin;
        xout[((long long)b * NJ + i) * Cout + 256 + c] =
            xin[((long long)b * NJ + i) * Cin + c];
    }
}

/* ---- 2-channel output head: h0/h1 GEMV then combine into d_out ---- */
__global__ void out_gemv(const float* __restrict__ X, const float* __restrict__ W,
                         float* __restrict__ h0, float* __restrict__ h1, int K)
{
    long long row = (long long)blockIdx.x * blockDim.x + threadIdx.x;
    if (row >= NROWS) return;
    const float* W0 = W;
    const float* W1 = W + (long long)K * 2;
    float a00 = 0, a01 = 0, a10 = 0, a11 = 0;
    const float* xp = X + row * K;
    for (int k = 0; k < K; ++k) {
        float v = xp[k];
        a00 += v * W0[k * 2];  a01 += v * W0[k * 2 + 1];
        a10 += v * W1[k * 2];  a11 += v * W1[k * 2 + 1];
    }
    h0[row * 2] = a00; h0[row * 2 + 1] = a01;
    h1[row * 2] = a10; h1[row * 2 + 1] = a11;
}

__global__ void out_combine(const float* __restrict__ h0, const float* __restrict__ h1,
                            const float* __restrict__ adj, const float* __restrict__ bias,
                            float* __restrict__ out)
{
    long long idx = (long long)blockIdx.x * blockDim.x + threadIdx.x;
    if (idx >= (long long)NROWS * 2) return;
    int d = (int)(idx & 1);
    long long row = idx >> 1;
    int i = (int)(row % NJ);
    long long rb = row - i;
    float v = adj[i * NJ + i] * h0[idx];
    int nb = c_nnb[i];
    for (int t = 0; t < nb; ++t) {
        int j = c_nbr[i][t];
        v += adj[i * NJ + j] * h1[(rb + j) * 2 + d];
    }
    out[idx] = v + bias[d];
}

/* =========================== host orchestration =========================== */

static inline long long cdiv(long long a, long long b) { return (a + b - 1) / b; }

struct Ws {
    float *A, *B, *C, *D;          /* 4 big activation buffers (RJ x 640) */
    float *adj, *scale, *shift, *psum, *psq;
    float *sBuf, *w1Buf, *wBuf, *hout;
};

static void run_gc(const float* Xin, float* Yout, float* H0, float* H1,
                   const float* W, const float* e, const float* bias,
                   const float* gamma, const float* beta,
                   int K, int N, const Ws& ws, hipStream_t st)
{
    adj_softmax_kernel<<<1, 32, 0, st>>>(e, ws.adj);
    dim3 grid(NROWS / 128, N / 64);
    wmma_dual_gemm<<<grid, 256, 0, st>>>(Xin, W, W + (size_t)K * N, H0, H1, K, N);
    long long total = (long long)NROWS * N;
    combine_kernel<<<cdiv(total, 256), 256, 0, st>>>(H0, H1, ws.adj, bias, Yout, N, total);
    bn_stats1<<<NROWS / 256, 256, 0, st>>>(Yout, ws.psum, ws.psq, N);
    bn_stats2<<<cdiv(N, 256), 256, 0, st>>>(ws.psum, ws.psq, gamma, beta, ws.scale, ws.shift, N);
    bn_apply<<<cdiv(total, 256), 256, 0, st>>>(Yout, ws.scale, ws.shift, N, total);
}

static void run_att(const float* Xres, const float* Yg2, float* Z, float* Xnext,
                    const float* f1w, const float* f1b, const float* f2w, const float* f2b,
                    int C, const Ws& ws, hipStream_t st)
{
    long long total = (long long)NROWS * C;
    int C4 = C / 4;
    add_kernel<<<cdiv(total, 256), 256, 0, st>>>(Xres, Yg2, Z, total);
    smean_kernel<<<cdiv((long long)NBATCH * C, 256), 256, 0, st>>>(Z, ws.sBuf, C);
    fc1_kernel<<<cdiv((long long)NBATCH * C4, 256), 256, 0, st>>>(ws.sBuf, f1w, f1b, ws.w1Buf, C, C4);
    fc2_kernel<<<cdiv((long long)NBATCH * NJ, 256), 256, 0, st>>>(ws.w1Buf, f2w, f2b, ws.wBuf, C4);
    att_apply<<<cdiv(total, 256), 256, 0, st>>>(Z, ws.wBuf, Xnext, C, total);
}

extern "C" void kernel_launch(void* const* d_in, const int* in_sizes, int n_in,
                              void* d_out, int out_size, void* d_ws, size_t ws_size,
                              hipStream_t stream)
{
    (void)in_sizes; (void)out_size;
    const float* x  = (const float*)d_in[0];
    const float* j1 = (const float*)d_in[1];
    const float* j2 = (const float*)d_in[2];
    const float* P[64];
    for (int i = 3; i < n_in && i < 64; ++i) P[i] = (const float*)d_in[i];

    /* param index map (setup_inputs insertion order, recursive) */
    enum {
        IN_W = 3, IN_E, IN_B, IN_G, IN_BT,                 /* input gc      */
        R1G1_W = 8,  R1G1_E, R1G1_B, R1G1_G, R1G1_BT,
        R1G2_W = 13, R1G2_E, R1G2_B, R1G2_G, R1G2_BT,
        R1_F1W = 18, R1_F1B, R1_F2W, R1_F2B,
        R2G1_W = 22, R2G1_E, R2G1_B, R2G1_G, R2G1_BT,
        R2G2_W = 27, R2G2_E, R2G2_B, R2G2_G, R2G2_BT,
        R2_F1W = 32, R2_F1B, R2_F2W, R2_F2B,
        R3G1_W = 36, R3G1_E, R3G1_B, R3G1_G, R3G1_BT,
        R3G2_W = 41, R3G2_E, R3G2_B, R3G2_G, R3G2_BT,
        R3_F1W = 46, R3_F1B, R3_F2W, R3_F2B,
        O1_W = 50, O1_E, O1_B,
        O2_W = 53, O2_E, O2_B
    };

    /* workspace layout */
    const size_t BIG = (size_t)NROWS * 640;
    float* w = (float*)d_ws;
    Ws ws;
    size_t off = 0;
    ws.A = w + off; off += BIG;
    ws.B = w + off; off += BIG;
    ws.C = w + off; off += BIG;
    ws.D = w + off; off += BIG;
    ws.adj   = w + off; off += NJ * NJ;
    ws.scale = w + off; off += 640;
    ws.shift = w + off; off += 640;
    ws.psum  = w + off; off += (size_t)336 * 640;
    ws.psq   = w + off; off += (size_t)336 * 640;
    ws.sBuf  = w + off; off += (size_t)NBATCH * 640;
    ws.w1Buf = w + off; off += (size_t)NBATCH * 160;
    ws.wBuf  = w + off; off += (size_t)NBATCH * NJ;
    ws.hout  = w + off; off += (size_t)NROWS * 4;
    if (ws_size < off * sizeof(float)) return;   /* insufficient scratch */

    float* out  = (float*)d_out;                 /* out1 then out2, each RJ*2 */
    float* out1 = out;
    float* out2 = out + (size_t)NROWS * 2;

    /* ---- input graph conv: (B,21,2) -> X128 in A ---- */
    {
        adj_softmax_kernel<<<1, 32, 0, stream>>>(P[IN_E], ws.adj);
        long long total = (long long)NROWS * 128;
        input_gemv_kernel<<<cdiv(total, 256), 256, 0, stream>>>(x, P[IN_W], ws.C, ws.D);
        combine_kernel<<<cdiv(total, 256), 256, 0, stream>>>(ws.C, ws.D, ws.adj, P[IN_B], ws.A, 128, total);
        bn_stats1<<<NROWS / 256, 256, 0, stream>>>(ws.A, ws.psum, ws.psq, 128);
        bn_stats2<<<1, 256, 0, stream>>>(ws.psum, ws.psq, P[IN_G], P[IN_BT], ws.scale, ws.shift, 128);
        bn_apply<<<cdiv(total, 256), 256, 0, stream>>>(ws.A, ws.scale, ws.shift, 128, total);
    }

    /* ---- res1 (128 -> 128): X in A ---- */
    run_gc(ws.A, ws.B, ws.C, ws.D, P[R1G1_W], P[R1G1_E], P[R1G1_B], P[R1G1_G], P[R1G1_BT], 128, 128, ws, stream);
    run_gc(ws.B, ws.C, ws.C, ws.D, P[R1G2_W], P[R1G2_E], P[R1G2_B], P[R1G2_G], P[R1G2_BT], 128, 128, ws, stream);
    run_att(ws.A, ws.C, ws.D, ws.A, P[R1_F1W], P[R1_F1B], P[R1_F2W], P[R1_F2B], 128, ws, stream);

    /* ---- res2 (concat j1 -> 384) ---- */
    concat_kernel<<<NBATCH, 256, 0, stream>>>(j1, ws.A, ws.B, 128, 384);
    run_gc(ws.B, ws.A, ws.C, ws.D, P[R2G1_W], P[R2G1_E], P[R2G1_B], P[R2G1_G], P[R2G1_BT], 384, 384, ws, stream);
    run_gc(ws.A, ws.C, ws.C, ws.D, P[R2G2_W], P[R2G2_E], P[R2G2_B], P[R2G2_G], P[R2G2_BT], 384, 384, ws, stream);
    run_att(ws.B, ws.C, ws.D, ws.A, P[R2_F1W], P[R2_F1B], P[R2_F2W], P[R2_F2B], 384, ws, stream);

    /* ---- out1 head: 384 -> 2 ---- */
    adj_softmax_kernel<<<1, 32, 0, stream>>>(P[O1_E], ws.adj);
    out_gemv<<<cdiv(NROWS, 256), 256, 0, stream>>>(ws.A, P[O1_W], ws.hout, ws.hout + (size_t)NROWS * 2, 384);
    out_combine<<<cdiv((long long)NROWS * 2, 256), 256, 0, stream>>>(
        ws.hout, ws.hout + (size_t)NROWS * 2, ws.adj, P[O1_B], out1);

    /* ---- res3 (concat j2 -> 640) ---- */
    concat_kernel<<<NBATCH, 256, 0, stream>>>(j2, ws.A, ws.B, 384, 640);
    run_gc(ws.B, ws.A, ws.C, ws.D, P[R3G1_W], P[R3G1_E], P[R3G1_B], P[R3G1_G], P[R3G1_BT], 640, 640, ws, stream);
    run_gc(ws.A, ws.C, ws.C, ws.D, P[R3G2_W], P[R3G2_E], P[R3G2_B], P[R3G2_G], P[R3G2_BT], 640, 640, ws, stream);
    run_att(ws.B, ws.C, ws.D, ws.A, P[R3_F1W], P[R3_F1B], P[R3_F2W], P[R3_F2B], 640, ws, stream);

    /* ---- out2 head: 640 -> 2 ---- */
    adj_softmax_kernel<<<1, 32, 0, stream>>>(P[O2_E], ws.adj);
    out_gemv<<<cdiv(NROWS, 256), 256, 0, stream>>>(ws.A, P[O2_W], ws.hout, ws.hout + (size_t)NROWS * 2, 640);
    out_combine<<<cdiv((long long)NROWS * 2, 256), 256, 0, stream>>>(
        ws.hout, ws.hout + (size_t)NROWS * 2, ws.adj, P[O2_B], out2);
}
